// CrossAttention_87393994539586
// MI455X (gfx1250) — matrix operand order repaired and checked by reading, not compiled, not run on hardware
//
#include <hip/hip_runtime.h>

// ---------------------------------------------------------------------------
// CDNA5 (gfx1250) cross-attention: WMMA projections (double-buffered LDS) +
// flash attention with TDM (tensor_load_to_lds) K-tile staging.
// ---------------------------------------------------------------------------

typedef _Float16 half_t;
typedef __attribute__((ext_vector_type(16))) _Float16     v16h;
typedef __attribute__((ext_vector_type(8)))  _Float16     v8h;
typedef __attribute__((ext_vector_type(8)))  float        v8f;
typedef __attribute__((ext_vector_type(4)))  unsigned int u32x4;
typedef __attribute__((ext_vector_type(8)))  int          i32x8;
typedef __attribute__((ext_vector_type(4)))  int          i32x4;

union H16 { v16h v; v8h h[2]; };

#define WMMA_F32_F16(a, b, c) \
  __builtin_amdgcn_wmma_f32_16x16x32_f16(false, (a), false, (b), (short)0, (c), false, false)

constexpr int   Bc    = 2;
constexpr int   Sc    = 2048;
constexpr int   Dc    = 1024;
constexpr int   Hc    = 16;
constexpr int   HDc   = 64;
constexpr int   Mrows = Bc * Sc;          // 4096
constexpr float SCALE = 0.125f;           // 64^-0.5
constexpr float L2E   = 1.4426950408889634f;

// ---------------------------------------------------------------------------
// GEMM with bias: C[M,N] = A[M,K] @ W[K,N] + bias[N]
// Block tile 128x64, BK=32, 8 waves -> 4x2 wave grid, each wave 32x32 (2x2 WMMA).
// Double-buffered LDS: next K-tile global loads overlap current tile's WMMAs.
// ---------------------------------------------------------------------------
template <typename TIn, typename TOut>
__global__ __launch_bounds__(256) void gemm_bias_wmma(
    const TIn* __restrict__ A, const float* __restrict__ W,
    const float* __restrict__ bias, TOut* __restrict__ C,
    int K, int N)
{
  constexpr int BM = 128, BN = 64, BK = 32, LDA = BK + 8, LDB = BK + 8;
  __shared__ half_t lA[2][BM * LDA];   // [m][k]
  __shared__ half_t lB[2][BN * LDB];   // [n][k]  (transposed W tile)

  const int  tid   = threadIdx.x;
  const int  lane  = tid & 31;
  const int  wid   = tid >> 5;
  const int  row16 = lane & 15;
  const bool hiL   = lane >= 16;
  const int  mWave = (wid >> 1) * 32;          // 0,32,64,96
  const int  nWave = (wid & 1) * 32;           // 0,32
  const int  mTile = blockIdx.y * BM;
  const int  nTile = blockIdx.x * BN;

  // per-thread staging assignments
  const int rA = tid >> 1, cbA = (tid & 1) * 16;      // A: 128 rows x 2 half-rows
  const int kB = tid >> 3, nbB = (tid & 7) * 8;       // W: 32 k-rows x 8 col-groups

  const TIn*   gA = A + (size_t)(mTile + rA) * K + cbA;
  const float* gW = W + (size_t)kB * N + nTile + nbB;

  v8h   aR0, aR1;
  float bR[8];

  auto loadTile = [&](int k0) {
    const TIn* g = gA + k0;
    #pragma unroll
    for (int e = 0; e < 8; ++e) aR0[e] = (half_t)(float)g[e];
    #pragma unroll
    for (int e = 0; e < 8; ++e) aR1[e] = (half_t)(float)g[e + 8];
    const float* gw = gW + (size_t)k0 * N;
    #pragma unroll
    for (int e = 0; e < 8; ++e) bR[e] = gw[e];
  };
  auto storeTile = [&](int buf) {
    *(v8h*)&lA[buf][rA * LDA + cbA]     = aR0;
    *(v8h*)&lA[buf][rA * LDA + cbA + 8] = aR1;
    #pragma unroll
    for (int e = 0; e < 8; ++e) lB[buf][(nbB + e) * LDB + kB] = (half_t)bR[e];
  };

  v8f acc[2][2] = {};

  loadTile(0);
  storeTile(0);
  __syncthreads();

  int buf = 0;
  for (int k0 = 0; k0 < K; k0 += BK) {
    const int nxt = k0 + BK;
    if (nxt < K) loadTile(nxt);            // global loads overlap WMMA below

    H16 aF[2], bF[2];
    #pragma unroll
    for (int mi = 0; mi < 2; ++mi) {
      const half_t* p = &lA[buf][(mWave + mi * 16 + row16) * LDA + (hiL ? 8 : 0)];
      aF[mi].h[0] = *(const v8h*)(p);
      aF[mi].h[1] = *(const v8h*)(p + 16);
    }
    #pragma unroll
    for (int ni = 0; ni < 2; ++ni) {
      const half_t* p = &lB[buf][(nWave + ni * 16 + row16) * LDB + (hiL ? 8 : 0)];
      bF[ni].h[0] = *(const v8h*)(p);
      bF[ni].h[1] = *(const v8h*)(p + 16);
    }
    #pragma unroll
    for (int mi = 0; mi < 2; ++mi)
      #pragma unroll
      for (int ni = 0; ni < 2; ++ni)
        acc[mi][ni] = WMMA_F32_F16(aF[mi].v, bF[ni].v, acc[mi][ni]);

    __syncthreads();
    if (nxt < K) storeTile(buf ^ 1);
    __syncthreads();
    buf ^= 1;
  }

  // ---- epilogue: C layout lane=col(N), vgpr r = row (+8 for hi half)
  const int gRow0 = mTile + mWave;
  const int gCol0 = nTile + nWave;
  #pragma unroll
  for (int ni = 0; ni < 2; ++ni) {
    const int   c  = gCol0 + ni * 16 + row16;
    const float bv = bias[c];
    #pragma unroll
    for (int mi = 0; mi < 2; ++mi) {
      const int r0 = gRow0 + mi * 16 + (hiL ? 8 : 0);
      #pragma unroll
      for (int r = 0; r < 8; ++r)
        C[(size_t)(r0 + r) * N + c] = (TOut)(acc[mi][ni][r] + bv);
    }
  }
}

// ---------------------------------------------------------------------------
// TDM: 2D tile load Global -> LDS, 2-byte elements, with LDS row padding of
// 4 DWORDs (8 halves) after every 32 DWORDs (64 halves) -> row stride 72.
// D# layout per CDNA5 ISA ch.8 (group0: count/lds/global/type, group1: dims).
// This toolchain's builtin takes 6 args: (g0, g1, g2, g3, g_extra, cpol).
// ---------------------------------------------------------------------------
#if __has_builtin(__builtin_amdgcn_tensor_load_to_lds)
#define HAVE_TDM 1
__device__ __forceinline__ void tdm_load_2d_f16(
    unsigned int lds_addr, const void* gptr,
    int dim0, int dim1, int tile0, int tile1, int stride0)
{
  const unsigned long long ga = (unsigned long long)(uintptr_t)gptr;
  u32x4 g0;
  g0[0] = 1u;                                           // count=1 (user D#)
  g0[1] = lds_addr;                                     // LDS byte address
  g0[2] = (unsigned int)(ga & 0xFFFFFFFFu);             // global_addr[31:0]
  g0[3] = (unsigned int)((ga >> 32) & 0x01FFFFFFu)      // global_addr[56:32]
        | (2u << 30);                                   // type = 2 ("image")
  i32x8 g1;
  g1[0] = (1 << 16)                                     // data_size = 2 bytes
        | (1 << 20)                                     // pad_enable
        | (4 << 22)                                     // pad_interval: 32 DW
        | (3 << 25);                                    // pad_amount: 4 DW
  g1[1] = (dim0 & 0xFFFF) << 16;                        // tensor_dim0[15:0]
  g1[2] = ((dim0 >> 16) & 0xFFFF) | ((dim1 & 0xFFFF) << 16);
  g1[3] = ((dim1 >> 16) & 0xFFFF) | ((tile0 & 0xFFFF) << 16);
  g1[4] = (tile1 & 0xFFFF);                             // tile_dim1 (tile_dim2=0)
  g1[5] = stride0;                                      // tensor_dim0_stride[31:0]
  g1[6] = 0;
  g1[7] = 0;
  const i32x4 z4 = {0, 0, 0, 0};
  const i32x8 z8 = {0, 0, 0, 0, 0, 0, 0, 0};
  __builtin_amdgcn_tensor_load_to_lds(g0, g1, z4, z4, z8, 0);
}
#else
#define HAVE_TDM 0
#endif

// ---------------------------------------------------------------------------
// Flash attention (causal). Block = 128 query rows of one (b,h); 8 waves each
// own 16 rows. K tile staged by the Tensor Data Mover (pad -> 72-half rows);
// V staged transposed by VALU; online softmax with shfl_xor row reductions.
// ---------------------------------------------------------------------------
__global__ __launch_bounds__(256) void flash_attn_wmma(
    const half_t* __restrict__ Q, const half_t* __restrict__ Kh,
    const half_t* __restrict__ V, half_t* __restrict__ O)
{
  constexpr int LDT = HDc + 8;                 // 72 (matches TDM pad)
  __shared__ half_t lK[64 * LDT];              // [key][d]   (TDM destination)
  __shared__ half_t lV[HDc * LDT];             // [d][key]   (transposed)
  __shared__ half_t lP[8 * 16 * LDT];          // per-wave 16x64 P tile

  const int    tid   = threadIdx.x;
  const int    lane  = tid & 31;
  const int    wid   = tid >> 5;
  const int    row16 = lane & 15;
  const bool   hiL   = lane >= 16;
  const int    b     = blockIdx.x / Hc;
  const int    h     = blockIdx.x % Hc;
  const int    qBase = blockIdx.y * 128;
  const int    qRow0 = qBase + wid * 16;
  const size_t bOff  = (size_t)b * Sc * Dc;
  const size_t head  = (size_t)h * HDc;

  // Q fragments held in registers for the whole kernel: 2 k-steps of 32 dims
  H16 qF[2];
  {
    const half_t* qp = Q + bOff + (size_t)(qRow0 + row16) * Dc + head;
    #pragma unroll
    for (int ks = 0; ks < 2; ++ks) {
      const half_t* p = qp + ks * 32 + (hiL ? 8 : 0);
      qF[ks].h[0] = *(const v8h*)(p);
      qF[ks].h[1] = *(const v8h*)(p + 16);
    }
  }

  float mRun[8], lRun[8];
  v8f   o[4] = {};
  #pragma unroll
  for (int r = 0; r < 8; ++r) { mRun[r] = -1.0e30f; lRun[r] = 0.0f; }

  half_t*   myP  = &lP[wid * 16 * LDT];
  const int jEnd = qBase + 127;                // causal upper bound for this block

  for (int j0 = 0; j0 <= jEnd; j0 += 64) {
    __syncthreads();                           // previous tile fully consumed
    // ---- stage K tile via TDM (one wave issues; uniform descriptor)
#if HAVE_TDM
    if (wid == 0) {
      tdm_load_2d_f16((unsigned int)(uintptr_t)&lK[0],
                      Kh + bOff + (size_t)j0 * Dc + head,
                      /*dim0*/ HDc, /*dim1*/ Sc - j0,
                      /*tile0*/ HDc, /*tile1*/ 64, /*stride0*/ Dc);
    }
#else
    {
      const int key = tid >> 2;
      const int db  = (tid & 3) * 16;
      const half_t* gk = Kh + bOff + (size_t)(j0 + key) * Dc + head + db;
      half_t*       sk = &lK[key * LDT + db];
      *(v8h*)(sk)     = *(const v8h*)(gk);
      *(v8h*)(sk + 8) = *(const v8h*)(gk + 8);
    }
#endif
    // ---- stage V transposed ([d][key]) with all threads (overlaps TDM)
    {
      const int key = tid >> 2;
      const int db  = (tid & 3) * 16;
      const half_t* gv = V + bOff + (size_t)(j0 + key) * Dc + head + db;
      #pragma unroll
      for (int e = 0; e < 16; ++e) lV[(db + e) * LDT + key] = gv[e];
    }
#if HAVE_TDM
    if (wid == 0) __builtin_amdgcn_s_wait_tensorcnt(0);
#endif
    __syncthreads();

    // ---- scores S = Q @ K^T  (4 key-subtiles x 2 k-steps)
    v8f s[4] = {};
    #pragma unroll
    for (int nt = 0; nt < 4; ++nt) {
      #pragma unroll
      for (int ks = 0; ks < 2; ++ks) {
        H16 kF;
        const half_t* p = &lK[(nt * 16 + row16) * LDT + ks * 32 + (hiL ? 8 : 0)];
        kF.h[0] = *(const v8h*)(p);
        kF.h[1] = *(const v8h*)(p + 16);
        s[nt] = WMMA_F32_F16(qF[ks].v, kF.v, s[nt]);
      }
    }

    // ---- causal mask + online softmax (rows live in 16-lane halves)
    #pragma unroll
    for (int r = 0; r < 8; ++r) {
      const int i  = qRow0 + r + (hiL ? 8 : 0);
      float     mx = -3.0e38f;
      #pragma unroll
      for (int nt = 0; nt < 4; ++nt) {
        const int j  = j0 + nt * 16 + row16;
        float     sv = (j <= i) ? s[nt][r] : -3.0e38f;
        s[nt][r] = sv;
        mx = fmaxf(mx, sv);
      }
      #pragma unroll
      for (int m = 8; m >= 1; m >>= 1) mx = fmaxf(mx, __shfl_xor(mx, m, 32));

      const float mNew  = fmaxf(mRun[r], mx);
      const float alpha = exp2f((mRun[r] - mNew) * (SCALE * L2E));
      mRun[r] = mNew;

      float sum = 0.0f;
      #pragma unroll
      for (int nt = 0; nt < 4; ++nt) {
        const float p = exp2f((s[nt][r] - mNew) * (SCALE * L2E));
        sum += p;
        myP[(r + (hiL ? 8 : 0)) * LDT + nt * 16 + row16] = (half_t)p;
      }
      #pragma unroll
      for (int m = 8; m >= 1; m >>= 1) sum += __shfl_xor(sum, m, 32);
      lRun[r] = lRun[r] * alpha + sum;
      #pragma unroll
      for (int nt = 0; nt < 4; ++nt) o[nt][r] = o[nt][r] * alpha;
    }

    // ---- O += P @ V  (P read back from per-wave LDS in A-layout)
    #pragma unroll
    for (int ks = 0; ks < 2; ++ks) {
      H16 pF;
      const half_t* pp = &myP[row16 * LDT + ks * 32 + (hiL ? 8 : 0)];
      pF.h[0] = *(const v8h*)(pp);
      pF.h[1] = *(const v8h*)(pp + 16);
      #pragma unroll
      for (int nt = 0; nt < 4; ++nt) {
        H16 vF;
        const half_t* vp = &lV[(nt * 16 + row16) * LDT + ks * 32 + (hiL ? 8 : 0)];
        vF.h[0] = *(const v8h*)(vp);
        vF.h[1] = *(const v8h*)(vp + 16);
        o[nt] = WMMA_F32_F16(pF.v, vF.v, o[nt]);
      }
    }
  }

  // ---- normalize and write attention output (f16, [B,S,D] layout)
  half_t* op = O + bOff + head;
  #pragma unroll
  for (int r = 0; r < 8; ++r) {
    const float inv  = 1.0f / lRun[r];
    const int   srow = qRow0 + r + (hiL ? 8 : 0);
    #pragma unroll
    for (int nt = 0; nt < 4; ++nt)
      op[(size_t)srow * Dc + nt * 16 + row16] = (half_t)(o[nt][r] * inv);
  }
}

// ---------------------------------------------------------------------------
extern "C" void kernel_launch(void* const* d_in, const int* in_sizes, int n_in,
                              void* d_out, int out_size, void* d_ws, size_t ws_size,
                              hipStream_t stream) {
  (void)in_sizes; (void)n_in; (void)out_size; (void)ws_size;

  const float* x  = (const float*)d_in[0];
  const float* y  = (const float*)d_in[1];
  const float* Wq = (const float*)d_in[2];
  const float* bq = (const float*)d_in[3];
  const float* Wk = (const float*)d_in[4];
  const float* bk = (const float*)d_in[5];
  const float* Wv = (const float*)d_in[6];
  const float* bv = (const float*)d_in[7];
  const float* Wo = (const float*)d_in[8];
  const float* bo = (const float*)d_in[9];
  float*       out = (float*)d_out;

  const size_t elems = (size_t)Mrows * Dc;       // 4Mi elements per tensor
  half_t* qh = (half_t*)d_ws;                    // 8 MiB each, 32 MiB total
  half_t* kh = qh + elems;
  half_t* vh = kh + elems;
  half_t* ah = vh + elems;

  const dim3 blk(256);
  const dim3 gProj(Dc / 64, Mrows / 128);        // 16 x 32
  const dim3 gAttn(Bc * Hc, Sc / 128);           // 32 x 16

  gemm_bias_wmma<float,  half_t><<<gProj, blk, 0, stream>>>(x,  Wq, bq, qh,  Dc, Dc);
  gemm_bias_wmma<float,  half_t><<<gProj, blk, 0, stream>>>(y,  Wk, bk, kh,  Dc, Dc);
  gemm_bias_wmma<float,  half_t><<<gProj, blk, 0, stream>>>(y,  Wv, bv, vh,  Dc, Dc);
  flash_attn_wmma<<<gAttn, blk, 0, stream>>>(qh, kh, vh, ah);
  gemm_bias_wmma<half_t, float ><<<gProj, blk, 0, stream>>>(ah, Wo, bo, out, Dc, Dc);
}
